// GraphAttentionLayer_44487271252604
// MI455X (gfx1250) — matrix operand order repaired and checked by reading, not compiled, not run on hardware
//
#include <hip/hip_runtime.h>
#include <hip/hip_bf16.h>

#define N_NODES 8192
#define IN_FEAT 256
#define OUT_FEAT 128
#define LRELU 0.2f

typedef __attribute__((ext_vector_type(2)))  float    v2f;
typedef __attribute__((ext_vector_type(4)))  float    v4f;
typedef __attribute__((ext_vector_type(8)))  float    v8f;
typedef __attribute__((ext_vector_type(4)))  int      v4i;
typedef __attribute__((ext_vector_type(8)))  __bf16   v8bf;
typedef __attribute__((ext_vector_type(16))) __bf16   v16bf;

// ---------------------------------------------------------------------------
// Kernel A: h = inputs @ W  (full f32 precision via V_WMMA_F32_16X16X4_F32)
// Writes hT (bf16, transposed [OUT_FEAT][N_NODES]) for the aggregation pass,
// plus s1 = h@a1, s2 = h@a2 (f32).
// Grid: 512 blocks x 256 threads (8 waves); wave w -> cols [16w,16w+16),
// block b -> rows [16b,16b+16).
// ---------------------------------------------------------------------------
__global__ __launch_bounds__(256) void gat_h_kernel(
    const float* __restrict__ inp, const float* __restrict__ W,
    const float* __restrict__ a,
    __bf16* __restrict__ hT,
    float* __restrict__ s1g, float* __restrict__ s2g)
{
    __shared__ float s1sh[16];
    __shared__ float s2sh[16];
    const int tid = threadIdx.x;
    if (tid < 16) { s1sh[tid] = 0.f; s2sh[tid] = 0.f; }
    __syncthreads();

    const int wave  = tid >> 5;
    const int lane  = tid & 31;
    const int row0  = blockIdx.x * 16;
    const int n0    = wave * 16;
    const int r     = lane & 15;       // A matrix row (M)
    const int khalf = lane >> 4;       // 0: K pair {0,1}, 1: K pair {2,3}
    const int n     = n0 + (lane & 15);

    const float* __restrict__ arow = inp + (size_t)(row0 + r) * IN_FEAT + 2 * khalf;
    const float* __restrict__ bcol = W + (size_t)(2 * khalf) * OUT_FEAT + n;

    v8f c = {};
    #pragma unroll 4
    for (int k = 0; k < IN_FEAT; k += 4) {
        v2f A; A.x = arow[k];            A.y = arow[k + 1];
        v2f B; B.x = bcol[k * OUT_FEAT]; B.y = bcol[(k + 1) * OUT_FEAT];
        c = __builtin_amdgcn_wmma_f32_16x16x4_f32(
                false, A, false, B, (short)0, c, false, false);
    }

    const float a1n = a[n];
    const float a2n = a[OUT_FEAT + n];

    #pragma unroll
    for (int v = 0; v < 8; ++v) {
        const int m = v + 8 * khalf;           // C layout: M = v + 8*(lane/16)
        const float hv = c[v];
        hT[(size_t)n * N_NODES + row0 + m] = (__bf16)hv;
        float p1 = hv * a1n;
        float p2 = hv * a2n;
        // reduce over the 16 lanes that share row m (distinct n).
        // xor offsets < 16 never cross the 16-lane halves -> full-width shfl ok
        #pragma unroll
        for (int off = 8; off; off >>= 1) {
            p1 += __shfl_xor(p1, off);
            p2 += __shfl_xor(p2, off);
        }
        if ((lane & 15) == 0) {
            atomicAdd(&s1sh[m], p1);
            atomicAdd(&s2sh[m], p2);
        }
    }
    __syncthreads();
    if (tid < 16) {
        s1g[row0 + tid] = s1sh[tid];
        s2g[row0 + tid] = s2sh[tid];
    }
}

// ---------------------------------------------------------------------------
// Kernel B: fused mask + unnormalized softmax + (attention @ h) + ELU.
// One pass over adj (268 MB, nontemporal). p = adj ? exp(lrelu(s1+s2)) : 0
// (no max-shift needed: |e| is O(8), exp cannot overflow f32).
// Grid: 256 blocks x 256 threads (8 waves); block -> 32 rows, wave -> 16 cols.
// 64-column K-tile per barrier (2 WMMA sub-steps), double-buffered in LDS;
// each wave runs 2 row-tiles sharing one B tile -> 4 WMMAs per barrier.
// ---------------------------------------------------------------------------
__global__ __launch_bounds__(256) void gat_attn_kernel(
    const int* __restrict__ adj,
    const __bf16* __restrict__ hT,
    const float* __restrict__ s1g, const float* __restrict__ s2g,
    float* __restrict__ out)
{
    __shared__ alignas(128) __bf16 psh[2][32][64];   // 8 KB p tiles
    __shared__ float s1sh[32];
    __shared__ float zsum[32];

    const int tid  = threadIdx.x;
    const int row0 = blockIdx.x * 32;
    if (tid < 32) { s1sh[tid] = s1g[row0 + tid]; zsum[tid] = 0.f; }
    __syncthreads();

    const int wave = tid >> 5;
    const int lane = tid & 31;

    // cooperative p-producer mapping: 32 rows x 64 cols, 8 cols per thread
    const int i  = tid >> 3;          // row in block 0..31 (fixed per thread)
    const int jl = (tid & 7) * 8;     // col offset 0,8,...,56
    const int* __restrict__ adjrow = adj + (size_t)(row0 + i) * N_NODES + jl;
    const float s1i = s1sh[i];
    float zacc = 0.f;

    // WMMA consumer mapping
    const int n     = wave * 16 + (lane & 15);  // output column / B column
    const int khalf = lane >> 4;                // K-half select
    const int ar    = lane & 15;                // A row
    const int abase = khalf * 8;                // A lo-chunk K offset
    const __bf16* __restrict__ bptr = hT + (size_t)n * N_NODES + khalf * 8;

    v8f c0 = {}, c1 = {};

    for (int kt = 0; kt < N_NODES / 64; ++kt) {
        const int buf = kt & 1;

        // ---- produce a 32x64 bf16 p tile (8 elements/thread) ----
        v4i av0 = __builtin_nontemporal_load((const v4i*)(adjrow + kt * 64));
        v4i av1 = __builtin_nontemporal_load((const v4i*)(adjrow + kt * 64 + 4));
        if (kt + 1 < N_NODES / 64)
            __builtin_prefetch(adjrow + (kt + 1) * 64, 0, 0);
        v4f s2a = *(const v4f*)(s2g + kt * 64 + jl);
        v4f s2b = *(const v4f*)(s2g + kt * 64 + jl + 4);

        v8bf pv;
        #pragma unroll
        for (int q = 0; q < 8; ++q) {
            float ev = s1i + (q < 4 ? s2a[q & 3] : s2b[q & 3]);
            ev = fmaxf(ev, LRELU * ev);                    // leaky-relu, slope<1
            int m = (q < 4 ? av0[q & 3] : av1[q & 3]);
            float p = (m > 0) ? __expf(ev) : 0.f;
            pv[q] = (__bf16)p;                             // hw cvt to bf16
            zacc += p;                                     // unbiased vs rounded
        }
        *(v8bf*)&psh[buf][i][jl] = pv;                     // single 16B DS store

        __syncthreads();   // one barrier/iter is safe with double buffering

        // ---- consume: 2 K sub-steps x (2 row-tiles sharing B) ----
        #pragma unroll
        for (int s = 0; s < 2; ++s) {
            const int kb = s * 32;
            v8bf a0lo = *(const v8bf*)&psh[buf][ar][kb + abase];
            v8bf a0hi = *(const v8bf*)&psh[buf][ar][kb + 16 + abase];
            v16bf A0 = __builtin_shufflevector(a0lo, a0hi,
                          0,1,2,3,4,5,6,7,8,9,10,11,12,13,14,15);
            v8bf a1lo = *(const v8bf*)&psh[buf][16 + ar][kb + abase];
            v8bf a1hi = *(const v8bf*)&psh[buf][16 + ar][kb + 16 + abase];
            v16bf A1 = __builtin_shufflevector(a1lo, a1hi,
                          0,1,2,3,4,5,6,7,8,9,10,11,12,13,14,15);
            v8bf blo = *(const v8bf*)(bptr + (size_t)kt * 64 + kb);
            v8bf bhi = *(const v8bf*)(bptr + (size_t)kt * 64 + kb + 16);
            v16bf B = __builtin_shufflevector(blo, bhi,
                          0,1,2,3,4,5,6,7,8,9,10,11,12,13,14,15);

            c0 = __builtin_amdgcn_wmma_f32_16x16x32_bf16(
                     false, A0, false, B, (short)0, c0, false, false);
            c1 = __builtin_amdgcn_wmma_f32_16x16x32_bf16(
                     false, A1, false, B, (short)0, c1, false, false);
        }
    }

    atomicAdd(&zsum[i], zacc);
    __syncthreads();

    #pragma unroll
    for (int v = 0; v < 8; ++v) {
        const int m = v + 8 * khalf;
        float h0 = c0[v] / zsum[m];
        float h1 = c1[v] / zsum[16 + m];
        h0 = h0 > 0.f ? h0 : (__expf(h0) - 1.f);   // ELU(alpha=1)
        h1 = h1 > 0.f ? h1 : (__expf(h1) - 1.f);
        out[(size_t)(row0 + m) * OUT_FEAT + n]      = h0;
        out[(size_t)(row0 + 16 + m) * OUT_FEAT + n] = h1;
    }
}

// ---------------------------------------------------------------------------
extern "C" void kernel_launch(void* const* d_in, const int* in_sizes, int n_in,
                              void* d_out, int out_size, void* d_ws, size_t ws_size,
                              hipStream_t stream)
{
    const float* inp = (const float*)d_in[0];   // (8192, 256) f32
    const int*   adj = (const int*)d_in[1];     // (8192, 8192) i32
    const float* W   = (const float*)d_in[2];   // (256, 128) f32
    const float* a   = (const float*)d_in[3];   // (256, 1) f32
    float* out = (float*)d_out;                 // (8192, 128) f32

    char* ws = (char*)d_ws;
    __bf16* hT = (__bf16*)ws;                                  // 128*8192*2 = 2 MiB
    float* s1 = (float*)(ws + 2u * 1024u * 1024u);             // 32 KiB
    float* s2 = (float*)(ws + 2u * 1024u * 1024u + 32768u);    // 32 KiB

    gat_h_kernel<<<N_NODES / 16, 256, 0, stream>>>(inp, W, a, hT, s1, s2);
    gat_attn_kernel<<<N_NODES / 32, 256, 0, stream>>>(adj, hT, s1, s2, out);
}
